// GraphTransformerLayer_12833362280653
// MI455X (gfx1250) — matrix operand order repaired and checked by reading, not compiled, not run on hardware
//
#include <hip/hip_runtime.h>
#include <hip/hip_bf16.h>

typedef __attribute__((ext_vector_type(16))) __bf16 v16bf;
typedef __attribute__((ext_vector_type(8)))  float  v8f;

#define DD 256    // feature dim
#define NN 512    // nodes
#define STRF 257  // fp32 LDS row stride (bank-conflict-free: 257*l % 64 distinct)
#define STRH 264  // bf16 LDS row stride (16B-aligned rows, 16-bank spread)

// ---------------------------------------------------------------- WMMA core
__device__ __forceinline__ v8f wmma_bf16(v16bf a, v16bf b, v8f c) {
  // D = A(16x32 bf16) x B(32x16 bf16) + C(16x16 f32)
  return __builtin_amdgcn_wmma_f32_16x16x32_bf16(false, a, false, b, (short)0, c,
                                                 false, false);
}

// B fragment: Wt is transposed bf16 weight [n][k]. Lane l (n = l&15):
// lanes 0-15 hold K = kb+{0..7, 16..23}, lanes 16-31 hold K = kb+{8..15, 24..31} (2 per VGPR).
__device__ __forceinline__ v16bf load_bfrag(const __bf16* Wt, int nb, int kb, int lane) {
  const __bf16* p = Wt + (nb + (lane & 15)) * DD + kb + ((lane & 16) ? 8 : 0);
  v16bf b;
#pragma unroll
  for (int t = 0; t < 16; ++t) b[t] = p[t < 8 ? t : t + 8];
  return b;
}

// A fragment from an fp32 row (ISA 16-bit A 16x32 layout):
// lanes 0-15: K = kb+{0..7, 16..23}; lanes 16-31: K = kb+{8..15, 24..31}
__device__ __forceinline__ v16bf afrag_f32(const float* row, int kb, int khalf) {
  v16bf a;
#pragma unroll
  for (int t = 0; t < 8; ++t) {
    a[t]     = (__bf16)row[kb + khalf + t];
    a[t + 8] = (__bf16)row[kb + 16 + khalf + t];
  }
  return a;
}

// A fragment from a bf16 row (same layout, no conversion).
__device__ __forceinline__ v16bf afrag_bf16(const __bf16* row, int kb, int khalf) {
  v16bf a;
#pragma unroll
  for (int t = 0; t < 8; ++t) {
    a[t]     = row[kb + khalf + t];
    a[t + 8] = row[kb + 16 + khalf + t];
  }
  return a;
}

// Per-wave LayerNorm over a 32x256 fp32 LDS tile (row stride STRF).
// Every lane reduces one row; stats broadcast via ds_bpermute (__shfl, wave32).
__device__ __forceinline__ void ln_wave32(float* A, const float* g, const float* be,
                                          int lane) {
  const float* r = A + lane * STRF;
  float s = 0.f, ss = 0.f;
  for (int c = 0; c < DD; ++c) { float v = r[c]; s += v; ss += v * v; }
  float mu = s * (1.0f / DD);
  float var = ss * (1.0f / DD) - mu * mu;
  float rstd = rsqrtf(var + 1e-5f);
  for (int idx = lane; idx < 32 * DD; idx += 32) {
    int row = idx >> 8, c = idx & 255;
    float m  = __shfl(mu, row, 32);
    float rs = __shfl(rstd, row, 32);
    A[row * STRF + c] = (A[row * STRF + c] - m) * rs * g[c] + be[c];
  }
}

// ---------------------------------------------------------------- weight prep
struct WSrc { const float* w[11]; };

// Wt[wi][n][k] = bf16(W[wi][k][n]) : transpose + convert (2.8 MB total, L2-hot)
__global__ void k_prep(WSrc p, __bf16* WtAll) {
  int wi  = blockIdx.y;
  int idx = blockIdx.x * 256 + threadIdx.x;     // 0..65535 ; idx = k*256 + n
  int k = idx >> 8, n = idx & 255;
  WtAll[wi * 65536 + n * DD + k] = (__bf16)p.w[wi][idx];
}

// ---------------------------------------------------------------- Q,K,V
// One wave = one 16-row tile of x; A-fragments kept in registers,
// three weight sets reuse them. (Tiny kernel: 0.1 GFLOP.)
__global__ void k_qkv(const float* x,
                      const __bf16* WqT, const __bf16* WkT, const __bf16* WvT,
                      const float* bq, const float* bk, const float* bv,
                      float* Q, float* K, float* V) {
  const int lane = threadIdx.x & 31, wave = threadIdx.x >> 5;
  const int rowbase = (blockIdx.x * 4 + wave) * 16;
  const int khalf = (lane & 16) ? 8 : 0;
  const int mrow0 = (lane & 16) ? 8 : 0;
  const int nloc  = lane & 15;
  const float* xr = x + (long)(rowbase + nloc) * DD;

  v16bf af[8];
#pragma unroll
  for (int kt = 0; kt < 8; ++kt) af[kt] = afrag_f32(xr, kt * 32, khalf);

  const __bf16* Wts[3] = {WqT, WkT, WvT};
  const float*  bs[3]  = {bq, bk, bv};
  float*        os[3]  = {Q, K, V};
#pragma unroll
  for (int s = 0; s < 3; ++s) {
    for (int nt = 0; nt < 16; ++nt) {
      const int nb = nt * 16, c = nb + nloc;
      v8f acc;
      const float bv0 = bs[s][c];
#pragma unroll
      for (int r = 0; r < 8; ++r) acc[r] = bv0;
#pragma unroll
      for (int kt = 0; kt < 8; ++kt)
        acc = wmma_bf16(af[kt], load_bfrag(Wts[s], nb, kt * 32, lane), acc);
#pragma unroll
      for (int r = 0; r < 8; ++r)
        os[s][(long)(rowbase + mrow0 + r) * DD + c] = acc[r];
    }
  }
}

// ---------------------------------------------------------------- edge gating
// One wave = 32 edge rows (same source node i, 32 consecutive j), two 16-row
// A-fragment sets so every B fragment feeds TWO WMMAs (halves L2 weight
// traffic). E1/E2 share the k-loop; Y = (Q_i*K_j/sqrt(DF))*(E1+1)+E2, bf16 out.
__global__ void k_edge_y(const float* e, const float* Q, const float* K,
                         const __bf16* WemT, const __bf16* WeaT,
                         const float* bem, const float* bea, __bf16* Y) {
  const int lane = threadIdx.x & 31, wave = threadIdx.x >> 5;
  const long base = ((long)blockIdx.x * 4 + wave) * 32; // edge base (i*512+jbase)
  const int i = (int)(base >> 9);
  const int jbase = (int)(base & 511);
  const int khalf = (lane & 16) ? 8 : 0;
  const int mrow0 = (lane & 16) ? 8 : 0;
  const int nloc  = lane & 15;
  const float* er0 = e + (base + nloc) * DD;
  const float* er1 = e + (base + 16 + nloc) * DD;

  v16bf af0[8], af1[8];
#pragma unroll
  for (int kt = 0; kt < 8; ++kt) {
    af0[kt] = afrag_f32(er0, kt * 32, khalf);
    af1[kt] = afrag_f32(er1, kt * 32, khalf);
  }

  const float rs = 0.17677669529663687f;       // 1/sqrt(32)
  for (int nt = 0; nt < 16; ++nt) {
    const int nb = nt * 16, c = nb + nloc;
    v8f m0, m1, a0, a1;
    const float b1 = bem[c], b2 = bea[c];
#pragma unroll
    for (int r = 0; r < 8; ++r) { m0[r] = b1; m1[r] = b1; a0[r] = b2; a1[r] = b2; }
#pragma unroll
    for (int kt = 0; kt < 8; ++kt) {
      v16bf bm = load_bfrag(WemT, nb, kt * 32, lane);
      m0 = wmma_bf16(af0[kt], bm, m0);
      m1 = wmma_bf16(af1[kt], bm, m1);
      v16bf ba = load_bfrag(WeaT, nb, kt * 32, lane);
      a0 = wmma_bf16(af0[kt], ba, a0);
      a1 = wmma_bf16(af1[kt], ba, a1);
    }
    const float q = Q[i * DD + c] * rs;
#pragma unroll
    for (int r = 0; r < 8; ++r) {
      const int j0 = jbase + mrow0 + r;
      const int j1 = j0 + 16;
      float y0 = q * K[j0 * DD + c] * (m0[r] + 1.0f) + a0[r];
      float y1 = q * K[j1 * DD + c] * (m1[r] + 1.0f) + a1[r];
      Y[(base + mrow0 + r) * DD + c]      = (__bf16)y0;
      Y[(base + 16 + mrow0 + r) * DD + c] = (__bf16)y1;
    }
  }
}

// ---------------------------------------------------------------- attention
// One block per query node i; thread = channel c. Online softmax over keys j
// fused with the attn*V accumulation (single coalesced pass over Y).
__global__ void k_attn(const __bf16* Y, const int* batch, const float* V,
                       float* wV) {
  const int i = blockIdx.x, c = threadIdx.x;
  const __bf16* Yr = Y + (long)i * NN * DD + c;
  const int* mr = batch + i * NN;
  float m = -3.0e38f, s = 0.f, acc = 0.f;
  for (int j = 0; j < NN; ++j) {
    float y = (mr[j] == 0) ? -9.0e9f : (float)Yr[(long)j * DD];
    float nm = fmaxf(m, y);
    float f = __expf(m - nm);
    float w = __expf(y - nm);
    s   = s * f + w;
    acc = acc * f + w * V[j * DD + c];
    m = nm;
  }
  wV[i * DD + c] = acc / s;
}

// ---------------------------------------------------------------- fused tails
// out = LN2( LN1(resid + Asrc@W0 + b0) + relu(LN1(..)@W1+b1)@W2 + b2 )
// EPATH=1: Asrc = Y (bf16, masked to -2 per edge row), resid = e.
// EPATH=0: Asrc = wV (fp32), resid = x.
// 32 rows per wave (two A-fragment sets per B fragment), 2 waves/block.
// A1: fp32 32x256 (stride 257), A2: bf16 32x256 (stride 264). GEMM2
// accumulates in-place into A1 (the e1/x1 residual), then LN2 + store.
template <int EPATH>
__global__ void k_tail(const void* Asrc, const int* batch, const float* resid,
                       const __bf16* W0T, const float* b0,
                       const __bf16* W1T, const float* b1,
                       const __bf16* W2T, const float* b2,
                       const float* g1, const float* be1,
                       const float* g2, const float* be2, float* out) {
  __shared__ float  act1[2][32 * STRF];
  __shared__ __bf16 act2[2][32 * STRH];
  const int lane = threadIdx.x & 31, wave = threadIdx.x >> 5;
  const long rowbase = ((long)blockIdx.x * 2 + wave) * 32;
  const int khalf = (lane & 16) ? 8 : 0;
  const int mrow0 = (lane & 16) ? 8 : 0;
  const int nloc  = lane & 15;
  float*  A1 = act1[wave];
  __bf16* A2 = act2[wave];

  { // residual into A1 (coalesced global read, padded LDS write)
    const float* r = resid + rowbase * DD;
    for (int idx = lane; idx < 32 * DD; idx += 32)
      A1[(idx >> 8) * STRF + (idx & 255)] = r[idx];
  }

  v16bf af0[8], af1[8];
  if (EPATH) {
    const __bf16* yr0 = (const __bf16*)Asrc + (rowbase + nloc) * DD;
    const __bf16* yr1 = (const __bf16*)Asrc + (rowbase + 16 + nloc) * DD;
    const bool m0 = (batch[rowbase + nloc] == 0);
    const bool m1 = (batch[rowbase + 16 + nloc] == 0);
    const __bf16 mval = (__bf16)(-2.0f);
#pragma unroll
    for (int kt = 0; kt < 8; ++kt) {
      v16bf a, b;
#pragma unroll
      for (int t = 0; t < 8; ++t) {
        a[t]     = m0 ? mval : yr0[kt * 32 + khalf + t];
        a[t + 8] = m0 ? mval : yr0[kt * 32 + 16 + khalf + t];
        b[t]     = m1 ? mval : yr1[kt * 32 + khalf + t];
        b[t + 8] = m1 ? mval : yr1[kt * 32 + 16 + khalf + t];
      }
      af0[kt] = a; af1[kt] = b;
    }
  } else {
    const float* wr0 = (const float*)Asrc + (rowbase + nloc) * DD;
    const float* wr1 = (const float*)Asrc + (rowbase + 16 + nloc) * DD;
#pragma unroll
    for (int kt = 0; kt < 8; ++kt) {
      af0[kt] = afrag_f32(wr0, kt * 32, khalf);
      af1[kt] = afrag_f32(wr1, kt * 32, khalf);
    }
  }

  // GEMM0: A1 += Asrc @ W0 + b0
  for (int nt = 0; nt < 16; ++nt) {
    const int nb = nt * 16, c = nb + nloc;
    v8f acc0, acc1; const float bv = b0[c];
#pragma unroll
    for (int r = 0; r < 8; ++r) { acc0[r] = bv; acc1[r] = bv; }
#pragma unroll
    for (int kt = 0; kt < 8; ++kt) {
      v16bf bf = load_bfrag(W0T, nb, kt * 32, lane);
      acc0 = wmma_bf16(af0[kt], bf, acc0);
      acc1 = wmma_bf16(af1[kt], bf, acc1);
    }
#pragma unroll
    for (int r = 0; r < 8; ++r) {
      A1[(mrow0 + r) * STRF + c]      += acc0[r];
      A1[(16 + mrow0 + r) * STRF + c] += acc1[r];
    }
  }
  ln_wave32(A1, g1, be1, lane);        // A1 = x1 / e1

  { // A-fragments from A1 (fp32 -> bf16)
    const float* r0 = A1 + nloc * STRF;
    const float* r1 = A1 + (16 + nloc) * STRF;
#pragma unroll
    for (int kt = 0; kt < 8; ++kt) {
      af0[kt] = afrag_f32(r0, kt * 32, khalf);
      af1[kt] = afrag_f32(r1, kt * 32, khalf);
    }
  }

  // GEMM1: A2 = bf16(relu(A1 @ W1 + b1))
  for (int nt = 0; nt < 16; ++nt) {
    const int nb = nt * 16, c = nb + nloc;
    v8f acc0, acc1; const float bv = b1[c];
#pragma unroll
    for (int r = 0; r < 8; ++r) { acc0[r] = bv; acc1[r] = bv; }
#pragma unroll
    for (int kt = 0; kt < 8; ++kt) {
      v16bf bf = load_bfrag(W1T, nb, kt * 32, lane);
      acc0 = wmma_bf16(af0[kt], bf, acc0);
      acc1 = wmma_bf16(af1[kt], bf, acc1);
    }
#pragma unroll
    for (int r = 0; r < 8; ++r) {
      A2[(mrow0 + r) * STRH + c]      = (__bf16)fmaxf(acc0[r], 0.f);
      A2[(16 + mrow0 + r) * STRH + c] = (__bf16)fmaxf(acc1[r], 0.f);
    }
  }

  { // A-fragments from A2 (already bf16; fully loaded before GEMM2 writes A1)
    const __bf16* r0 = A2 + nloc * STRH;
    const __bf16* r1 = A2 + (16 + nloc) * STRH;
#pragma unroll
    for (int kt = 0; kt < 8; ++kt) {
      af0[kt] = afrag_bf16(r0, kt * 32, khalf);
      af1[kt] = afrag_bf16(r1, kt * 32, khalf);
    }
  }

  // GEMM2 (in-place): A1 = A1 + relu(..) @ W2 + b2
  for (int nt = 0; nt < 16; ++nt) {
    const int nb = nt * 16, c = nb + nloc;
    v8f acc0, acc1; const float bv = b2[c];
#pragma unroll
    for (int r = 0; r < 8; ++r) { acc0[r] = bv; acc1[r] = bv; }
#pragma unroll
    for (int kt = 0; kt < 8; ++kt) {
      v16bf bf = load_bfrag(W2T, nb, kt * 32, lane);
      acc0 = wmma_bf16(af0[kt], bf, acc0);
      acc1 = wmma_bf16(af1[kt], bf, acc1);
    }
#pragma unroll
    for (int r = 0; r < 8; ++r) {
      A1[(mrow0 + r) * STRF + c]      += acc0[r];
      A1[(16 + mrow0 + r) * STRF + c] += acc1[r];
    }
  }
  ln_wave32(A1, g2, be2, lane);

  { // coalesced store
    float* o = out + rowbase * DD;
    for (int idx = lane; idx < 32 * DD; idx += 32)
      o[idx] = A1[(idx >> 8) * STRF + (idx & 255)];
  }
}

// ---------------------------------------------------------------- launcher
extern "C" void kernel_launch(void* const* d_in, const int* in_sizes, int n_in,
                              void* d_out, int out_size, void* d_ws,
                              size_t ws_size, hipStream_t stream) {
  const float* x     = (const float*)d_in[0];
  const float* e     = (const float*)d_in[1];
  const int*   batch = (const int*)d_in[2];
  const float* W[11]; const float* B[11];
  for (int i = 0; i < 11; ++i) {
    W[i] = (const float*)d_in[3 + 2 * i];
    B[i] = (const float*)d_in[4 + 2 * i];
  }
  const float* gX1 = (const float*)d_in[25]; const float* beX1 = (const float*)d_in[26];
  const float* gX2 = (const float*)d_in[27]; const float* beX2 = (const float*)d_in[28];
  const float* gE1 = (const float*)d_in[29]; const float* beE1 = (const float*)d_in[30];
  const float* gE2 = (const float*)d_in[31]; const float* beE2 = (const float*)d_in[32];

  // workspace carve-up  (~131.4 MB total)
  char* w = (char*)d_ws;
  size_t off = 0;
  __bf16* WtAll = (__bf16*)(w + off); off += 11ull * 65536 * 2;
  float* Q   = (float*)(w + off); off += (size_t)NN * DD * 4;
  float* Km  = (float*)(w + off); off += (size_t)NN * DD * 4;
  float* V   = (float*)(w + off); off += (size_t)NN * DD * 4;
  float* wV  = (float*)(w + off); off += (size_t)NN * DD * 4;
  __bf16* Y  = (__bf16*)(w + off); off += (size_t)NN * NN * DD * 2;
  if (ws_size < off) return;  // workspace too small for this plan

  float* Xout = (float*)d_out;            // [512,256]
  float* Eout = Xout + (size_t)NN * DD;   // [512,512,256]

  // weight index: 0 Wq,1 Wk,2 Wv,3 Wem,4 Wea,5 Wxo,6 Weo,7 WX1,8 WX2,9 WE1,10 WE2
  WSrc P;
  for (int i = 0; i < 11; ++i) P.w[i] = W[i];
  k_prep<<<dim3(256, 11), 256, 0, stream>>>(P, WtAll);

  k_qkv<<<8, 128, 0, stream>>>(x, WtAll, WtAll + 65536, WtAll + 2 * 65536,
                               B[0], B[1], B[2], Q, Km, V);

  // 262144 edges / 32 per wave / 4 waves per block = 2048 blocks
  k_edge_y<<<2048, 128, 0, stream>>>(e, Q, Km, WtAll + 3 * 65536,
                                     WtAll + 4 * 65536, B[3], B[4], Y);

  k_attn<<<NN, DD, 0, stream>>>(Y, batch, V, wV);

  // 512 rows / 32 per wave / 2 waves per block = 8 blocks
  k_tail<0><<<8, 64, 0, stream>>>(wV, nullptr, x,
                                  WtAll + 5 * 65536, B[5],
                                  WtAll + 7 * 65536, B[7],
                                  WtAll + 8 * 65536, B[8],
                                  gX1, beX1, gX2, beX2, Xout);

  // 262144 rows / 32 per wave / 2 waves per block = 4096 blocks
  k_tail<1><<<4096, 64, 0, stream>>>(Y, batch, e,
                                     WtAll + 6 * 65536, B[6],
                                     WtAll + 9 * 65536, B[9],
                                     WtAll + 10 * 65536, B[10],
                                     gE1, beE1, gE2, beE2, Eout);
}
